// LinearAttention_31086973288776
// MI455X (gfx1250) — matrix operand order repaired and checked by reading, not compiled, not run on hardware
//
#include <hip/hip_runtime.h>
#include <hip/hip_bf16.h>
#include <math.h>

#define D_MODEL 1024
#define NUM_HEADS 16
#define HEAD_DIM 64
#define BATCH 4
#define SEQ 4096
#define NROWS (BATCH * SEQ) /* 16384 */
#define KSTAGES (D_MODEL / 64) /* 16 */

typedef _Float16 half_t;
typedef _Float16 v8h  __attribute__((ext_vector_type(8)));
typedef _Float16 v16h __attribute__((ext_vector_type(16)));
typedef float    v8f  __attribute__((ext_vector_type(8)));
typedef unsigned int u32x4 __attribute__((ext_vector_type(4)));
typedef int          i32x4 __attribute__((ext_vector_type(4)));
typedef int          i32x8 __attribute__((ext_vector_type(8)));

__device__ __forceinline__ v16h cat16(v8h lo, v8h hi) {
  return __builtin_shufflevector(lo, hi, 0,1,2,3,4,5,6,7,8,9,10,11,12,13,14,15);
}
// A-matrix 16x32 f16 fragment: lane holds row M=lane&15; g=lane>>4 selects the
// K interleave. halves 0..7 -> K = 8g..8g+7 ; halves 8..15 -> K = 16+8g..23+8g
__device__ __forceinline__ v16h ldA(const half_t* rp, int g) {
  v8h lo = *(const v8h*)(rp + 8 * g);
  v8h hi = *(const v8h*)(rp + 16 + 8 * g);
  return cat16(lo, hi);
}
// B-matrix 32x16 f16 fragment from [N][K] storage: lane holds col N=lane&15,
// halves 0..7 -> K = 8g..8g+7 ; halves 8..15 -> K = 16+8g..23+8g
__device__ __forceinline__ v16h ldB(const half_t* rp, int g) {
  v8h lo = *(const v8h*)(rp + 8 * g);
  v8h hi = *(const v8h*)(rp + 16 + 8 * g);
  return cat16(lo, hi);
}
__device__ __forceinline__ v8f wmma_f16(v16h a, v16h b, v8f c) {
  return __builtin_amdgcn_wmma_f32_16x16x32_f16(false, a, false, b, (short)0, c,
                                                false, false);
}
__device__ __forceinline__ v8f zero8() {
  v8f z = {0.f, 0.f, 0.f, 0.f, 0.f, 0.f, 0.f, 0.f};
  return z;
}
__device__ __forceinline__ float feat(float v) { // elu(v)+1
  return (v > 0.f) ? (v + 1.f) : __expf(v);
}

// ---------------------------------------------------------------------------
// Tensor Data Mover: DMA one 128-row x 64-half f16 tile (row stride
// `stride_elems` halves) into LDS at byte offset lds_off, with 4 DWORDs of LDS
// padding after every 32 DWORDs (one row) -> landed row stride 72 halves.
// D# per CDNA5 ISA 8.3/8.4. Tracked with TENSORcnt; in-order per wave.
// ---------------------------------------------------------------------------
__device__ __forceinline__ void tdm_load_tile_f16(const half_t* gsrc,
                                                  unsigned int lds_off,
                                                  unsigned int stride_elems) {
  unsigned long long ga = (unsigned long long)(uintptr_t)gsrc;
  u32x4 g0;
  g0[0] = 1u;                                   // count=1, user mode
  g0[1] = lds_off;                              // lds_addr (bytes)
  g0[2] = (unsigned int)ga;                     // global_addr[31:0]
  g0[3] = ((unsigned int)(ga >> 32) & 0x1FFFFFFu) | (2u << 30); // hi, type=2
  i32x8 g1;
  g1[0] = (1 << 16)                              // data_size = 2 bytes
        | (1 << 20)                              // pad_enable
        | (4 << 22)                              // pad_interval: 32 DWORDs
        | (3 << 25);                             // pad_amount: 4 DWORDs
  g1[1] = (64 << 16);                            // tensor_dim0 = 64 (lo16)
  g1[2] = (128 << 16);                           // dim0 hi=0 | tensor_dim1=128
  g1[3] = (64 << 16);                            // dim1 hi=0 | tile_dim0 = 64
  g1[4] = 128;                                   // tile_dim1=128, tile_dim2=0
  g1[5] = (int)stride_elems;                     // tensor_dim0_stride lo32
  g1[6] = 0;
  g1[7] = 0;
  i32x4 z4 = {0, 0, 0, 0};
#if defined(__clang_major__) && __clang_major__ >= 23
  i32x8 z8 = {0, 0, 0, 0, 0, 0, 0, 0};
  __builtin_amdgcn_tensor_load_to_lds(g0, g1, z4, z4, z8, 0);
#else
  __builtin_amdgcn_tensor_load_to_lds(g0, g1, z4, z4, 0);
#endif
}

// ---------------------------------------------------------------------------
// One-time fp32 -> f16 conversion of a 1024x1024 weight matrix.
// ---------------------------------------------------------------------------
__global__ __launch_bounds__(256) void cvt_w_kernel(const float* __restrict__ in,
                                                    half_t* __restrict__ out) {
  const size_t i = ((size_t)blockIdx.x * 256 + threadIdx.x) * 8;
  const float4* p = (const float4*)(in + i);
  float4 f0 = p[0], f1 = p[1];
  v8h h = {(half_t)f0.x, (half_t)f0.y, (half_t)f0.z, (half_t)f0.w,
           (half_t)f1.x, (half_t)f1.y, (half_t)f1.z, (half_t)f1.w};
  *(v8h*)(out + i) = h;
}

// ---------------------------------------------------------------------------
// Normal-orientation projection (used for K and V): C[row][col] = X @ W^T + b.
// A = X (fp32, staged with fused v_cvt_f16_f32), B = W f16 (double-buffered
// TDM). Output written TRANSPOSED f16 [bh][d][l]; with C-layout M=r+8g the 8
// accumulator elements are 8 consecutive l at fixed d -> one v8h store each.
// ---------------------------------------------------------------------------
template <bool ACT>
__global__ __launch_bounds__(256) void proj_nt(const float* __restrict__ X,
                                               const half_t* __restrict__ Wh,
                                               const float* __restrict__ bias,
                                               half_t* __restrict__ Outt) {
  __shared__ __align__(16) half_t As[2][128][72];
  __shared__ __align__(16) half_t Bs[2][128][72];
  const int tid = threadIdx.x, lane = tid & 31, wave = tid >> 5;
  const int g = lane >> 4, ln = lane & 15;
  const int m_base = blockIdx.x * 128; // rows
  const int n_base = blockIdx.y * 128; // cols
  const int wm = (wave & 1) * 64, wn = (wave >> 1) * 32;

  v8f c[4][2];
  for (int i = 0; i < 4; i++)
    for (int j = 0; j < 2; j++) c[i][j] = zero8();

  if (wave == 0)
    tdm_load_tile_f16(Wh + (size_t)n_base * D_MODEL,
                      (unsigned int)(uintptr_t)&Bs[0][0][0], D_MODEL);

  for (int s = 0; s < KSTAGES; s++) {
    const int cur = s & 1, nxt = cur ^ 1;
    const int k0 = s * 64;
    if (wave == 0) {
      if (s + 1 < KSTAGES) {
        tdm_load_tile_f16(Wh + (size_t)n_base * D_MODEL + k0 + 64,
                          (unsigned int)(uintptr_t)&Bs[nxt][0][0], D_MODEL);
        __builtin_amdgcn_s_wait_tensorcnt(1); // stage s landed, s+1 in flight
      } else {
        __builtin_amdgcn_s_wait_tensorcnt(0);
      }
    }
    { // fp32 -> f16 staged A tile
      const int r = tid >> 1, co = (tid & 1) * 32;
      const float* ap = X + (size_t)(m_base + r) * D_MODEL + k0 + co;
      const float4* av = (const float4*)ap;
      for (int i = 0; i < 4; i++) {
        float4 f0 = av[2 * i], f1 = av[2 * i + 1];
        v8h h = {(half_t)f0.x, (half_t)f0.y, (half_t)f0.z, (half_t)f0.w,
                 (half_t)f1.x, (half_t)f1.y, (half_t)f1.z, (half_t)f1.w};
        *(v8h*)&As[cur][r][co + 8 * i] = h;
      }
    }
    __syncthreads();
    for (int kk = 0; kk < 64; kk += 32) {
      v16h a[4], b[2];
      for (int mt = 0; mt < 4; mt++)
        a[mt] = ldA(&As[cur][wm + 16 * mt + ln][kk], g);
      for (int nt = 0; nt < 2; nt++)
        b[nt] = ldB(&Bs[cur][wn + 16 * nt + ln][kk], g);
      for (int mt = 0; mt < 4; mt++)
        for (int nt = 0; nt < 2; nt++)
          c[mt][nt] = wmma_f16(a[mt], b[nt], c[mt][nt]);
    }
    __syncthreads();
  }

  const int bb = m_base >> 12; // whole block is one batch index
  for (int nt = 0; nt < 2; nt++) {
    const int col = n_base + wn + 16 * nt + ln;
    const float bv = bias[col];
    const int h = col >> 6, d = col & 63;
    half_t* orow = Outt + ((size_t)((bb * NUM_HEADS + h) * HEAD_DIM + d)) * SEQ;
    for (int mt = 0; mt < 4; mt++) {
      const int l = (m_base & 4095) + wm + 16 * mt + 8 * g;
      v8h hv;
      for (int r = 0; r < 8; r++) {
        float v = c[mt][nt][r] + bv;
        if (ACT) v = feat(v);
        hv[r] = (half_t)v;
      }
      *(v8h*)(orow + l) = hv; // 8 consecutive l, 16B store
    }
  }
}

// ---------------------------------------------------------------------------
// Flipped-orientation projection (used for Q and the output projection):
// C[o][n] = (W @ X^T). A = W f16 (TDM, double-buffered); B = X rows
// (fp32-staged for Q, TDM f16 for the output projection from ctx).
// Epilogue: o varies within a lane -> contiguous stores:
//   MODE 0: f16 v8h into Q [bh][l][d] (+ bias + elu+1)
//   MODE 3: fp32 2x float4 into d_out row-major (+ bias)
// ---------------------------------------------------------------------------
template <int MODE>
__global__ __launch_bounds__(256) void proj_tn(const void* __restrict__ Bin,
                                               const half_t* __restrict__ Wh,
                                               const float* __restrict__ bias,
                                               void* __restrict__ Out) {
  constexpr bool ACT = (MODE == 0);
  constexpr bool BTDM = (MODE == 3);
  constexpr short NL = BTDM ? 2 : 1; // TDM loads per stage
  __shared__ __align__(16) half_t Wls[2][128][72];
  __shared__ __align__(16) half_t Xls[2][128][72];
  const int tid = threadIdx.x, lane = tid & 31, wave = tid >> 5;
  const int g = lane >> 4, ln = lane & 15;
  const int o_base = blockIdx.x * 128; // output-channel tile (M dim)
  const int n_base = blockIdx.y * 128; // row tile (N dim)
  const int wm = (wave & 1) * 64, wn = (wave >> 1) * 32;

  v8f c[4][2];
  for (int i = 0; i < 4; i++)
    for (int j = 0; j < 2; j++) c[i][j] = zero8();

  if (wave == 0) {
    tdm_load_tile_f16(Wh + (size_t)o_base * D_MODEL,
                      (unsigned int)(uintptr_t)&Wls[0][0][0], D_MODEL);
    if (BTDM)
      tdm_load_tile_f16((const half_t*)Bin + (size_t)n_base * D_MODEL,
                        (unsigned int)(uintptr_t)&Xls[0][0][0], D_MODEL);
  }

  for (int s = 0; s < KSTAGES; s++) {
    const int cur = s & 1, nxt = cur ^ 1;
    const int k0 = s * 64;
    if (wave == 0) {
      if (s + 1 < KSTAGES) {
        tdm_load_tile_f16(Wh + (size_t)o_base * D_MODEL + k0 + 64,
                          (unsigned int)(uintptr_t)&Wls[nxt][0][0], D_MODEL);
        if (BTDM)
          tdm_load_tile_f16((const half_t*)Bin + (size_t)n_base * D_MODEL + k0 + 64,
                            (unsigned int)(uintptr_t)&Xls[nxt][0][0], D_MODEL);
        __builtin_amdgcn_s_wait_tensorcnt(NL); // stage s landed
      } else {
        __builtin_amdgcn_s_wait_tensorcnt(0);
      }
    }
    if (!BTDM) { // fp32 -> f16 staged B tile (rows of X)
      const int r = tid >> 1, co = (tid & 1) * 32;
      const float* ap =
          (const float*)Bin + (size_t)(n_base + r) * D_MODEL + k0 + co;
      const float4* av = (const float4*)ap;
      for (int i = 0; i < 4; i++) {
        float4 f0 = av[2 * i], f1 = av[2 * i + 1];
        v8h h = {(half_t)f0.x, (half_t)f0.y, (half_t)f0.z, (half_t)f0.w,
                 (half_t)f1.x, (half_t)f1.y, (half_t)f1.z, (half_t)f1.w};
        *(v8h*)&Xls[cur][r][co + 8 * i] = h;
      }
    }
    __syncthreads();
    for (int kk = 0; kk < 64; kk += 32) {
      v16h a[4], b[2];
      for (int mt = 0; mt < 4; mt++)
        a[mt] = ldA(&Wls[cur][wm + 16 * mt + ln][kk], g);
      for (int nt = 0; nt < 2; nt++)
        b[nt] = ldB(&Xls[cur][wn + 16 * nt + ln][kk], g);
      for (int mt = 0; mt < 4; mt++)
        for (int nt = 0; nt < 2; nt++)
          c[mt][nt] = wmma_f16(a[mt], b[nt], c[mt][nt]);
    }
    __syncthreads();
  }

  for (int nt = 0; nt < 2; nt++) {
    const int n = n_base + wn + 16 * nt + ln; // global row, fixed per lane
    for (int mt = 0; mt < 4; mt++) {
      const int o8 = o_base + wm + 16 * mt + 8 * g; // 8 consecutive channels
      const float4 bv0 = *(const float4*)(bias + o8);
      const float4 bv1 = *(const float4*)(bias + o8 + 4);
      const float bvv[8] = {bv0.x, bv0.y, bv0.z, bv0.w,
                            bv1.x, bv1.y, bv1.z, bv1.w};
      if (MODE == 0) {
        const int bb = n >> 12, l = n & 4095;
        const int h = o8 >> 6, d = o8 & 63;
        v8h hv;
        for (int r = 0; r < 8; r++)
          hv[r] = (half_t)feat(c[mt][nt][r] + bvv[r]);
        *(v8h*)((half_t*)Out +
                ((size_t)(bb * NUM_HEADS + h) * SEQ + l) * HEAD_DIM + d) = hv;
      } else {
        __align__(16) float vo[8];
        for (int r = 0; r < 8; r++) vo[r] = c[mt][nt][r] + bvv[r];
        float* dst = (float*)Out + (size_t)n * D_MODEL + o8;
        *(float4*)(dst) = *(float4*)&vo[0];
        *(float4*)(dst + 4) = *(float4*)&vo[4];
      }
    }
  }
}

// ---------------------------------------------------------------------------
// Per-(b,h): KV[d][e] = sum_l K[l,d]*V[l,e] (WMMA over L), Ksum[d] = sum_l K.
// Cross-wave reduction through LDS ds_add_f32 atomics. KV written transposed
// f16 ([e][d]) so it is A-major for the flipped ctx GEMM.
// ---------------------------------------------------------------------------
__global__ __launch_bounds__(256) void kv_ksum_kernel(
    const half_t* __restrict__ Kt, const half_t* __restrict__ Vt,
    half_t* __restrict__ KVt, float* __restrict__ Ksum) {
  __shared__ __align__(16) float kvs[64][64];
  __shared__ float ksp[4][64];
  const int bh = blockIdx.x;
  const int tid = threadIdx.x, lane = tid & 31, wave = tid >> 5;
  const int g = lane >> 4, ln = lane & 15;
  const half_t* KtB = Kt + (size_t)bh * HEAD_DIM * SEQ;
  const half_t* VtB = Vt + (size_t)bh * HEAD_DIM * SEQ;

  for (int i = tid; i < 64 * 64; i += 256) ((float*)kvs)[i] = 0.f;
  __syncthreads();

  v8f c[4][4];
  for (int i = 0; i < 4; i++)
    for (int j = 0; j < 4; j++) c[i][j] = zero8();

  const int lbeg = wave * (SEQ / 8);
  for (int l0 = lbeg; l0 < lbeg + SEQ / 8; l0 += 32) {
    v16h a[4], b[4];
    for (int dt = 0; dt < 4; dt++)
      a[dt] = ldA(KtB + (size_t)(16 * dt + ln) * SEQ + l0, g); // A = K^T [d][l]
    for (int et = 0; et < 4; et++)
      b[et] = ldB(VtB + (size_t)(16 * et + ln) * SEQ + l0, g); // B from V^T
    for (int dt = 0; dt < 4; dt++)
      for (int et = 0; et < 4; et++)
        c[dt][et] = wmma_f16(a[dt], b[et], c[dt][et]);
  }
  for (int dt = 0; dt < 4; dt++)
    for (int et = 0; et < 4; et++)
      for (int r = 0; r < 8; r++)
        atomicAdd(&kvs[16 * dt + r + 8 * g][16 * et + ln], c[dt][et][r]);

  { // Ksum partials: thread -> (d = tid&63, slice = tid>>6)
    const int d = tid & 63, s = tid >> 6;
    const half_t* kp = KtB + (size_t)d * SEQ + s * (SEQ / 4);
    float acc = 0.f;
    for (int i = 0; i < SEQ / 4; i += 8) {
      v8h v = *(const v8h*)(kp + i);
      for (int j = 0; j < 8; j++) acc += (float)v[j];
    }
    ksp[s][d] = acc;
  }
  __syncthreads();

  for (int i = tid; i < 64 * 64; i += 256) {
    const int d = i >> 6, e = i & 63;
    KVt[(size_t)bh * 4096 + (size_t)e * 64 + d] = (half_t)kvs[d][e];
  }
  if (tid < 64)
    Ksum[(size_t)bh * 64 + tid] =
        ksp[0][tid] + ksp[1][tid] + ksp[2][tid] + ksp[3][tid];
}

// ---------------------------------------------------------------------------
// Z[bh][l] = 1 / (dot(Q[bh,l,:], Ksum[bh,:]) + 1e-6)
// ---------------------------------------------------------------------------
__global__ __launch_bounds__(256) void z_kernel(const half_t* __restrict__ Qh,
                                                const float* __restrict__ Ksum,
                                                float* __restrict__ Z) {
  __shared__ float ks[64];
  const int tid = threadIdx.x;
  const size_t idx = (size_t)blockIdx.x * 256 + tid; // = bh*4096 + l
  const int bh = (int)(idx >> 12);
  if (tid < 64) ks[tid] = Ksum[(size_t)bh * 64 + tid];
  __syncthreads();
  const half_t* q = Qh + idx * HEAD_DIM;
  float s = 0.f;
  for (int i = 0; i < 8; i++) {
    v8h v = *(const v8h*)(q + 8 * i);
    for (int j = 0; j < 8; j++) s += (float)v[j] * ks[8 * i + j];
  }
  Z[idx] = 1.f / (s + 1e-6f);
}

// ---------------------------------------------------------------------------
// Flipped ctx GEMM: C[e][l] = KV^T @ Q^T restricted to one (b,h);
// A = KVt [e][d] (contiguous), B = Q rows [l][d] (contiguous). Epilogue:
// e varies within a lane -> v8h store into ctx[b,l,h*64+e]; one Z load/lane.
// ---------------------------------------------------------------------------
__global__ __launch_bounds__(256) void ctx_kernel(const half_t* __restrict__ Qh,
                                                  const half_t* __restrict__ KVt,
                                                  const float* __restrict__ Z,
                                                  half_t* __restrict__ Ctx) {
  const int bh = blockIdx.x >> 4; // 64 batch*heads
  const int lt = blockIdx.x & 15; // 16 l-tiles of 256
  const int tid = threadIdx.x, lane = tid & 31, wave = tid >> 5;
  const int g = lane >> 4, ln = lane & 15;
  const int bb = bh >> 4, h = bh & 15;
  const half_t* QB = Qh + (size_t)bh * SEQ * HEAD_DIM;
  const half_t* KVB = KVt + (size_t)bh * 4096;
  const int l0 = lt * 256 + wave * 32;

  v8f c[4][2]; // [e-tile][l-tile]
  for (int i = 0; i < 4; i++)
    for (int j = 0; j < 2; j++) c[i][j] = zero8();

  for (int k0 = 0; k0 < HEAD_DIM; k0 += 32) {
    v16h a[4], b[2];
    for (int mt = 0; mt < 4; mt++)
      a[mt] = ldA(KVB + (size_t)(16 * mt + ln) * 64 + k0, g); // KV^T [e][d]
    for (int nt = 0; nt < 2; nt++)
      b[nt] = ldB(QB + (size_t)(l0 + 16 * nt + ln) * HEAD_DIM + k0, g);
    for (int mt = 0; mt < 4; mt++)
      for (int nt = 0; nt < 2; nt++)
        c[mt][nt] = wmma_f16(a[mt], b[nt], c[mt][nt]);
  }
  for (int nt = 0; nt < 2; nt++) {
    const int l = l0 + 16 * nt + ln; // fixed per lane
    const float z = Z[(size_t)bh * SEQ + l];
    half_t* dst = Ctx + ((size_t)bb * SEQ + l) * D_MODEL + h * HEAD_DIM;
    for (int mt = 0; mt < 4; mt++) {
      const int e8 = 16 * mt + 8 * g; // 8 consecutive e
      v8h hv;
      for (int r = 0; r < 8; r++) hv[r] = (half_t)(c[mt][nt][r] * z);
      *(v8h*)(dst + e8) = hv;
    }
  }
}

extern "C" void kernel_launch(void* const* d_in, const int* in_sizes, int n_in,
                              void* d_out, int out_size, void* d_ws,
                              size_t ws_size, hipStream_t stream) {
  const float* q  = (const float*)d_in[0];
  const float* k  = (const float*)d_in[1];
  const float* v  = (const float*)d_in[2];
  const float* wq = (const float*)d_in[3];
  const float* bq = (const float*)d_in[4];
  const float* wk = (const float*)d_in[5];
  const float* bk = (const float*)d_in[6];
  const float* wv = (const float*)d_in[7];
  const float* bv = (const float*)d_in[8];
  const float* wo = (const float*)d_in[9];
  const float* bo = (const float*)d_in[10];

  char* ws = (char*)d_ws;
  half_t* Qh  = (half_t*)ws; ws += (size_t)64 * SEQ * HEAD_DIM * 2; // 33.5 MB
  half_t* Kt  = (half_t*)ws; ws += (size_t)64 * SEQ * HEAD_DIM * 2;
  half_t* Vt  = (half_t*)ws; ws += (size_t)64 * SEQ * HEAD_DIM * 2;
  half_t* KVt = (half_t*)ws; ws += (size_t)64 * 64 * 64 * 2;
  float*  Ks  = (float*)ws;  ws += (size_t)64 * 64 * 4;
  float*  Zb  = (float*)ws;  ws += (size_t)64 * SEQ * 4;
  half_t* Ctx = (half_t*)ws; ws += (size_t)NROWS * D_MODEL * 2;
  half_t* Wqh = (half_t*)ws; ws += (size_t)D_MODEL * D_MODEL * 2;
  half_t* Wkh = (half_t*)ws; ws += (size_t)D_MODEL * D_MODEL * 2;
  half_t* Wvh = (half_t*)ws; ws += (size_t)D_MODEL * D_MODEL * 2;
  half_t* Woh = (half_t*)ws; ws += (size_t)D_MODEL * D_MODEL * 2;

  const int cvt_blocks = (D_MODEL * D_MODEL) / (256 * 8); // 512
  cvt_w_kernel<<<cvt_blocks, 256, 0, stream>>>(wq, Wqh);
  cvt_w_kernel<<<cvt_blocks, 256, 0, stream>>>(wk, Wkh);
  cvt_w_kernel<<<cvt_blocks, 256, 0, stream>>>(wv, Wvh);
  cvt_w_kernel<<<cvt_blocks, 256, 0, stream>>>(wo, Woh);

  dim3 block(256);
  dim3 grid_nt(NROWS / 128, D_MODEL / 128); // rows x cols
  dim3 grid_tn(D_MODEL / 128, NROWS / 128); // cols x rows
  proj_tn<0><<<grid_tn, block, 0, stream>>>(q, Wqh, bq, Qh);
  proj_nt<true><<<grid_nt, block, 0, stream>>>(k, Wkh, bk, Kt);
  proj_nt<false><<<grid_nt, block, 0, stream>>>(v, Wvh, bv, Vt);
  kv_ksum_kernel<<<64, 256, 0, stream>>>(Kt, Vt, KVt, Ks);
  z_kernel<<<(64 * SEQ) / 256, 256, 0, stream>>>(Qh, Ks, Zb);
  ctx_kernel<<<64 * 16, 256, 0, stream>>>(Qh, KVt, Zb, Ctx);
  proj_tn<3><<<grid_tn, block, 0, stream>>>(Ctx, Woh, bo, d_out);
}
